// OptiMetal2BNoResidual_15848429322419
// MI455X (gfx1250) — matrix operand order repaired and checked by reading, not compiled, not run on hardware
//
#include <hip/hip_runtime.h>
#include <hip/hip_bf16.h>
#include <math.h>
#include <stdint.h>

// ---------------------------------------------------------------------------
// Types for CDNA5 WMMA (wave32, v_wmma_f32_16x16x32_bf16)
// ---------------------------------------------------------------------------
typedef __bf16 bf16_t;
typedef __attribute__((ext_vector_type(16))) __bf16 v16bf;
typedef __attribute__((ext_vector_type(8)))  __bf16 v8bf;
typedef __attribute__((ext_vector_type(8)))  float  v8f;

#define HEADS 4
#define HC    64
#define HIDD  256
#define NBASIS 64
#define RMAXF 5.5f
#define PI_F  3.14159265358979323846f

__device__ __forceinline__ bf16_t f2bf(float f) {
  unsigned u = __float_as_uint(f);
  unsigned r = u + 0x7fffu + ((u >> 16) & 1u);   // round-to-nearest-even
  unsigned short h = (unsigned short)(r >> 16);
  return __builtin_bit_cast(bf16_t, h);
}
__device__ __forceinline__ float bf2f(bf16_t b) {
  unsigned short h = __builtin_bit_cast(unsigned short, b);
  return __uint_as_float(((unsigned)h) << 16);
}
// float atomic max, valid with -inf init (signed max for >=0, unsigned min for <0)
__device__ __forceinline__ void atomicMaxF(float* addr, float v) {
  if (v >= 0.f) atomicMax((int*)addr, __float_as_int(v));
  else          atomicMin((unsigned int*)addr, __float_as_uint(v));
}

// CDNA5 async global->LDS copy (ASYNCcnt-tracked) + wait
__device__ __forceinline__ void async_ld_b128(unsigned lds_off, unsigned long long gaddr) {
  asm volatile("global_load_async_to_lds_b128 %0, %1, off"
               :: "v"(lds_off), "v"(gaddr) : "memory");
}
__device__ __forceinline__ void wait_async0() {
  asm volatile("s_wait_asynccnt 0x0" ::: "memory");
}

// ---------------------------------------------------------------------------
// WMMA GEMM:  C[M,Nreal] = act(A[M,K](bf16) @ Wpacked[K,Npad](bf16) + bias)
// A row-major bf16, ldA=K (K multiple of 64, M multiple of 64).
// Wpacked: Wp[((k>>4)*Npad + n)*16 + (k&15)] -> B-fragment (lane = column n,
//   lane-half h = K 8h..8h+7 and 16+8h..16+8h+7) is two b128 per lane, one per k16 block.
// Block = 256 threads = 8 waves = 64 rows x 256 cols.
//   wave: rowHalf = wave&1 (32 rows, 2 accumulator row-tiles), colGrp = wave>>1.
// B strip staged through LDS in KC=64 chunks with CDNA5 async loads,
// double buffered; all 4 B fragments of a K-step preloaded into distinct
// registers so the 8 WMMAs issue back-to-back behind one s_wait_dscnt.
// ---------------------------------------------------------------------------
#define KC 64

template<int ACT>
__global__ __launch_bounds__(256)
void k_wmma_gemm(const bf16_t* __restrict__ A, const bf16_t* __restrict__ Bp,
                 const float* __restrict__ bias, int M, int K, int Nreal, int Npad,
                 float* __restrict__ outF, bf16_t* __restrict__ outB)
{
  __shared__ bf16_t smem[2][KC * 256];       // 2 x 32KB

  const int tid  = threadIdx.x;
  const int lane = tid & 31;
  const int wave = tid >> 5;
  const int l16  = lane & 15;
  const int half = lane >> 4;
  const int rowHalf = wave & 1;
  const int colGrp  = wave >> 1;
  const size_t cb = (size_t)blockIdx.y * 256;                    // block col base
  const size_t rowBase = (size_t)blockIdx.x * 64 + (size_t)rowHalf * 32;

  const bf16_t* arow0 = A + (rowBase + 0  + (size_t)l16) * (size_t)K;
  const bf16_t* arow1 = A + (rowBase + 16 + (size_t)l16) * (size_t)K;

  v8f acc[2][4] = {};

  const int nChunks = K / KC;
  const unsigned long long bpBase = (unsigned long long)(uintptr_t)Bp;
  const unsigned ldsBase0 = (unsigned)(uintptr_t)(void*)&smem[0][0];
  const unsigned ldsBase1 = (unsigned)(uintptr_t)(void*)&smem[1][0];

  // stage one KC x 256 chunk of packed B into LDS buffer `buf` (async)
  auto stage = [&](int c, int buf) {
#pragma unroll
    for (int p = 0; p < 8; ++p) {
      const unsigned o = (unsigned)(p * 256 + tid) * 16u;        // byte in chunk [0,32768)
      const unsigned seg = o >> 13, within = o & 8191u;          // 4 x 8KB segments
      const unsigned long long g =
          bpBase + ((unsigned long long)(c * 4 + seg) * (unsigned long long)Npad
                    + (unsigned long long)cb) * 32ull + within;
      async_ld_b128((buf ? ldsBase1 : ldsBase0) + o, g);
    }
  };

  stage(0, 0);
  for (int c = 0; c < nChunks; ++c) {
    wait_async0();
    __syncthreads();                         // chunk c visible to all waves
    if (c + 1 < nChunks) stage(c + 1, (c + 1) & 1);
    const bf16_t* sb = smem[c & 1];
    const int kc0 = c * KC;
#pragma unroll
    for (int ks = 0; ks < KC; ks += 32) {
      // A fragments for both row tiles (global, 2x b128 each)
      union { v16bf v; v8bf h[2]; } a0, a1;
      a0.h[0] = *(const v8bf*)(arow0 + kc0 + ks + half * 8);
      a0.h[1] = *(const v8bf*)(arow0 + kc0 + ks + 16 + half * 8);
      a1.h[0] = *(const v8bf*)(arow1 + kc0 + ks + half * 8);
      a1.h[1] = *(const v8bf*)(arow1 + kc0 + ks + 16 + half * 8);
      // All 4 B fragments of this K-step into distinct registers (LDS)
      const int kb = (ks >> 4);              // first k16-block of this K-step
      union { v16bf v; v8bf h[2]; } bfv[4];
#pragma unroll
      for (int t = 0; t < 4; ++t) {
        const int n = colGrp * 64 + t * 16 + l16;
        const bf16_t* bp = sb + (size_t)kb * (256 * 16) + (size_t)n * 16 + half * 8;
        bfv[t].h[0] = *(const v8bf*)(bp);
        bfv[t].h[1] = *(const v8bf*)(bp + 256 * 16);
      }
      // 8 back-to-back WMMAs
#pragma unroll
      for (int t = 0; t < 4; ++t) {
        acc[0][t] = __builtin_amdgcn_wmma_f32_16x16x32_bf16(
            false, a0.v, false, bfv[t].v, (short)0, acc[0][t], false, false);
        acc[1][t] = __builtin_amdgcn_wmma_f32_16x16x32_bf16(
            false, a1.v, false, bfv[t].v, (short)0, acc[1][t], false, false);
      }
    }
  }

#pragma unroll
  for (int rt = 0; rt < 2; ++rt) {
#pragma unroll
    for (int t = 0; t < 4; ++t) {
      const size_t col = cb + (size_t)colGrp * 64 + (size_t)t * 16 + (size_t)l16;
      if (col < (size_t)Nreal) {
        const float bia = bias ? bias[col] : 0.f;
#pragma unroll
        for (int v = 0; v < 8; ++v) {
          const size_t r = rowBase + (size_t)rt * 16 + (size_t)v + (size_t)half * 8;
          float val = acc[rt][t][v] + bia;
          if (ACT == 1) val = fmaxf(val, 0.f);
          if (outF) outF[r * (size_t)Nreal + col] = val;
          if (outB) outB[r * (size_t)Nreal + col] = f2bf(val);
        }
      }
    }
  }
}

// ---------------------------------------------------------------------------
// Weight packing f32 [K,Nn] -> bf16 packed [K,Npad] (zero padded columns)
// ---------------------------------------------------------------------------
__global__ void k_pack_w(const float* __restrict__ W, bf16_t* __restrict__ out,
                         int K, int Nn, int Npad) {
  int tid = blockIdx.x * blockDim.x + threadIdx.x;
  if (tid >= K * Npad) return;
  int k = tid / Npad, n = tid % Npad;
  float v = (n < Nn) ? W[(size_t)k * Nn + n] : 0.f;
  out[((size_t)(k >> 4) * Npad + n) * 16 + (k & 15)] = f2bf(v);
}

__global__ void k_fill(float* __restrict__ p, float v, int n) {
  int tid = blockIdx.x * blockDim.x + threadIdx.x;
  if (tid < n) p[tid] = v;
}
__global__ void k_f32_to_bf16(const float* __restrict__ in, bf16_t* __restrict__ out, int n) {
  int tid = blockIdx.x * blockDim.x + threadIdx.x;
  if (tid < n) out[tid] = f2bf(in[tid]);
}

// ---------------------------------------------------------------------------
// Atom embedding gather -> bf16 [N,64]
// ---------------------------------------------------------------------------
__global__ void k_embed(const int* __restrict__ an, const float* __restrict__ emb,
                        bf16_t* __restrict__ out, int N) {
  int tid = blockIdx.x * blockDim.x + threadIdx.x;
  if (tid >= N * 64) return;
  int n = tid >> 6, c = tid & 63;
  out[tid] = f2bf(emb[(size_t)an[n] * 64 + c]);
}

// ---------------------------------------------------------------------------
// Edge geometry + gaussian basis * cosine envelope -> bf16 [E,64]
// ---------------------------------------------------------------------------
__global__ void k_edge_geom(const float* __restrict__ pos, const float* __restrict__ lat,
                            const int* __restrict__ ei, const float* __restrict__ pbc,
                            const int* __restrict__ ebat, bf16_t* __restrict__ out, int E) {
  int tid = blockIdx.x * blockDim.x + threadIdx.x;
  if (tid >= E * NBASIS) return;
  int e = tid >> 6, j = tid & 63;
  int s = ei[e], d = ei[E + e], b = ebat[e];
  const float* L = lat + (size_t)b * 9;
  float px = pbc[e * 3 + 0], py = pbc[e * 3 + 1], pz = pbc[e * 3 + 2];
  float dd = 0.f;
#pragma unroll
  for (int c = 0; c < 3; ++c) {
    float ev = px * L[0 * 3 + c] + py * L[1 * 3 + c] + pz * L[2 * 3 + c]
             + pos[(size_t)d * 3 + c] - pos[(size_t)s * 3 + c];
    dd += ev * ev;
  }
  dd = sqrtf(dd);
  float cj = RMAXF * (float)j / (float)(NBASIS - 1);
  float r = dd - cj;
  float rbf = expf(-r * r * 0.125f);                         // 2*WIDTH^2 = 8
  float env = (dd < RMAXF) ? 0.5f * (cosf(PI_F * dd / RMAXF) + 1.f) : 0.f;
  out[(size_t)e * NBASIS + j] = f2bf(rbf * env);
}

// ---------------------------------------------------------------------------
// GATv2: logits + segment max / exp-sum / aggregate / combine
// ---------------------------------------------------------------------------
__global__ void k_gat_logit(const float* __restrict__ xl, const float* __restrict__ xr,
                            const bf16_t* __restrict__ ee, const int* __restrict__ ei,
                            const float* __restrict__ att, float* __restrict__ logit,
                            float* __restrict__ mx, int E) {
  int tid = blockIdx.x * blockDim.x + threadIdx.x;
  if (tid >= E * HEADS) return;
  int e = tid >> 2, h = tid & 3;
  int s = ei[e], d = ei[E + e];
  const float*  pl = xl + (size_t)s * HIDD + h * HC;
  const float*  pr = xr + (size_t)d * HIDD + h * HC;
  const bf16_t* pe = ee + (size_t)e * HIDD + h * HC;
  const float*  pa = att + h * HC;
  float acc = 0.f;
  for (int c = 0; c < HC; ++c) {
    float z = pl[c] + pr[c] + bf2f(pe[c]);
    z = (z > 0.f) ? z : 0.2f * z;                            // LeakyReLU(0.2)
    acc += pa[c] * z;
  }
  logit[tid] = acc;
  atomicMaxF(&mx[(size_t)d * HEADS + h], acc);
}

__global__ void k_gat_expsum(const float* __restrict__ logit, const float* __restrict__ mx,
                             const int* __restrict__ ei, float* __restrict__ pex,
                             float* __restrict__ den, int E) {
  int tid = blockIdx.x * blockDim.x + threadIdx.x;
  if (tid >= E * HEADS) return;
  int e = tid >> 2, h = tid & 3;
  int d = ei[E + e];
  float p = expf(logit[tid] - mx[(size_t)d * HEADS + h]);
  pex[tid] = p;
  atomicAdd(&den[(size_t)d * HEADS + h], p);
}

__global__ void k_gat_aggr(const float* __restrict__ pex, const float* __restrict__ den,
                           const float* __restrict__ xl, const int* __restrict__ ei,
                           float* __restrict__ agg, int E) {
  int tid = blockIdx.x * blockDim.x + threadIdx.x;
  if (tid >= E * HIDD) return;
  int e = tid >> 8, c = tid & 255, h = c >> 6;
  int s = ei[e], d = ei[E + e];
  float alpha = pex[(size_t)e * HEADS + h] / (den[(size_t)d * HEADS + h] + 1e-16f);
  atomicAdd(&agg[(size_t)d * HIDD + c], alpha * xl[(size_t)s * HIDD + c]);
}

__global__ void k_gat_combine(const float* __restrict__ agg, const float* __restrict__ gbias,
                              const float* __restrict__ xres, float* __restrict__ xoutF,
                              bf16_t* __restrict__ xoutB, int N) {
  int tid = blockIdx.x * blockDim.x + threadIdx.x;
  if (tid >= N * HIDD) return;
  int c = tid & 255;
  float v = agg[tid] + gbias[c] + xres[tid];
  xoutF[tid] = v;
  xoutB[tid] = f2bf(v);
}

// ---------------------------------------------------------------------------
// Vector-attention pooling (per-channel segment softmax over graphs)
// ---------------------------------------------------------------------------
__global__ void k_pool_max(const float* __restrict__ g, const int* __restrict__ nbat,
                           float* __restrict__ gm, int N) {
  int tid = blockIdx.x * blockDim.x + threadIdx.x;
  if (tid >= N * HIDD) return;
  int n = tid >> 8, c = tid & 255;
  atomicMaxF(&gm[(size_t)nbat[n] * HIDD + c], g[tid]);
}
__global__ void k_pool_expsum(const float* __restrict__ g, const float* __restrict__ gm,
                              const int* __restrict__ nbat, float* __restrict__ gp,
                              float* __restrict__ gd, int N) {
  int tid = blockIdx.x * blockDim.x + threadIdx.x;
  if (tid >= N * HIDD) return;
  int n = tid >> 8, c = tid & 255;
  float p = expf(g[tid] - gm[(size_t)nbat[n] * HIDD + c]);
  gp[tid] = p;
  atomicAdd(&gd[(size_t)nbat[n] * HIDD + c], p);
}
__global__ void k_pool_weighted(const float* __restrict__ gp, const float* __restrict__ gd,
                                const float* __restrict__ x, const int* __restrict__ nbat,
                                float* __restrict__ pooled, int N) {
  int tid = blockIdx.x * blockDim.x + threadIdx.x;
  if (tid >= N * HIDD) return;
  int n = tid >> 8, c = tid & 255;
  size_t bi = (size_t)nbat[n] * HIDD + c;
  atomicAdd(&pooled[bi], gp[tid] / (gd[bi] + 1e-16f) * x[tid]);
}

// ---------------------------------------------------------------------------
// Host-side launcher
// ---------------------------------------------------------------------------
static inline void launch_gemm(hipStream_t s, const bf16_t* A, const bf16_t* Bp,
                               const float* bias, int M, int K, int Nr, int Np,
                               float* oF, bf16_t* oB, int act) {
  dim3 grid((unsigned)(M / 64), (unsigned)(Np / 256));
  if (act) k_wmma_gemm<1><<<grid, 256, 0, s>>>(A, Bp, bias, M, K, Nr, Np, oF, oB);
  else     k_wmma_gemm<0><<<grid, 256, 0, s>>>(A, Bp, bias, M, K, Nr, Np, oF, oB);
}
static inline unsigned cdiv(long long a, int b) { return (unsigned)((a + b - 1) / b); }

#define NW 23   // number of weight matrices to pack

extern "C" void kernel_launch(void* const* d_in, const int* in_sizes, int n_in,
                              void* d_out, int out_size, void* d_ws, size_t ws_size,
                              hipStream_t stream) {
  (void)n_in; (void)out_size; (void)ws_size;
  const int N = in_sizes[0];           // 16384 nodes
  const int E = in_sizes[6];           // 262144 edges
  const int G = in_sizes[2] / 9;       // 256 graphs

  const int*   anum = (const int*)  d_in[0];
  const float* pos  = (const float*)d_in[1];
  const float* lat  = (const float*)d_in[2];
  const int*   ei   = (const int*)  d_in[3];
  const float* pbc  = (const float*)d_in[4];
  const int*   nbat = (const int*)  d_in[5];
  const int*   ebat = (const int*)  d_in[6];
  const float* aemb = (const float*)d_in[7];

  // params flatten order (setup_inputs insertion order):
  // 8..13 node_mlp, 14..19 edge_mlp, 20..25 gat0 (Wl,Wr,We,att,bias,Wres),
  // 26..31 gat1, 32..37 pool_mlp, 38..43 spectra_mlp, 44..49 drude_mlp
  struct WS { int idx, K, Nn, Np; };
  const WS wsp[NW] = {
    {8,64,256,256},{10,256,256,256},{12,256,256,256},          // node_mlp
    {14,64,256,256},{16,256,256,256},{18,256,256,256},         // edge_mlp
    {20,256,256,256},{21,256,256,256},{22,256,256,256},{25,256,256,256}, // gat0
    {26,256,256,256},{27,256,256,256},{28,256,256,256},{31,256,256,256}, // gat1
    {32,256,256,256},{34,256,256,256},{36,256,256,256},        // pool_mlp
    {38,256,2048,2048},{40,2048,2048,2048},{42,2048,4002,4096},// spectra_mlp
    {44,256,256,256},{46,256,256,256},{48,256,1,256},          // drude_mlp
  };

  size_t off = 0;
  auto carve = [&](size_t bytes) -> void* {
    void* p = (char*)d_ws + off;
    off += (bytes + 255) & ~(size_t)255;
    return p;
  };
  bf16_t* wp[NW];
  for (int i = 0; i < NW; ++i)
    wp[i] = (bf16_t*)carve((size_t)wsp[i].K * wsp[i].Np * sizeof(bf16_t));
  auto Wp = [&](int idx) -> bf16_t* {
    for (int i = 0; i < NW; ++i) if (wsp[i].idx == idx) return wp[i];
    return nullptr;
  };
  auto Bf = [&](int idx) -> const float* { return (const float*)d_in[idx]; };

  bf16_t* x0B    = (bf16_t*)carve((size_t)N * 64  * 2);
  bf16_t* bufN1  = (bf16_t*)carve((size_t)N * 256 * 2);
  bf16_t* bufN2  = (bf16_t*)carve((size_t)N * 256 * 2);
  bf16_t* xB     = (bf16_t*)carve((size_t)N * 256 * 2);
  float*  xFa    = (float*) carve((size_t)N * 256 * 4);
  float*  xFb    = (float*) carve((size_t)N * 256 * 4);
  float*  xlF    = (float*) carve((size_t)N * 256 * 4);
  float*  xrF    = (float*) carve((size_t)N * 256 * 4);
  float*  xresF  = (float*) carve((size_t)N * 256 * 4);
  float*  aggF   = (float*) carve((size_t)N * 256 * 4);
  float*  gF     = (float*) carve((size_t)N * 256 * 4);
  float*  gpF    = (float*) carve((size_t)N * 256 * 4);
  bf16_t* eAttrB = (bf16_t*)carve((size_t)E * 64  * 2);
  bf16_t* ebufA  = (bf16_t*)carve((size_t)E * 256 * 2);  // h1, later ee
  bf16_t* ebufB  = (bf16_t*)carve((size_t)E * 256 * 2);  // h2
  bf16_t* eaB    = (bf16_t*)carve((size_t)E * 256 * 2);  // edge features (persist)
  float*  logitF = (float*) carve((size_t)E * 4 * 4);
  float*  pexF   = (float*) carve((size_t)E * 4 * 4);
  float*  mF     = (float*) carve((size_t)N * 4 * 4);
  float*  denF   = (float*) carve((size_t)N * 4 * 4);
  float*  gmF    = (float*) carve((size_t)G * 256 * 4);
  float*  gdF    = (float*) carve((size_t)G * 256 * 4);
  float*  pooledF= (float*) carve((size_t)G * 256 * 4);
  bf16_t* pooledB= (bf16_t*)carve((size_t)G * 256 * 2);
  bf16_t* sH1    = (bf16_t*)carve((size_t)G * 2048 * 2);
  bf16_t* sH2    = (bf16_t*)carve((size_t)G * 2048 * 2);

  // ---- pack all weight matrices to WMMA-B-friendly bf16 layout ----
  for (int i = 0; i < NW; ++i) {
    int total = wsp[i].K * wsp[i].Np;
    k_pack_w<<<cdiv(total, 256), 256, 0, stream>>>(Bf(wsp[i].idx), wp[i],
                                                   wsp[i].K, wsp[i].Nn, wsp[i].Np);
  }

  // ---- node embedding + node_mlp ----
  k_embed<<<cdiv((long long)N * 64, 256), 256, 0, stream>>>(anum, aemb, x0B, N);
  launch_gemm(stream, x0B,   Wp(8),  Bf(9),  N, 64,  256, 256, nullptr, bufN1, 1);
  launch_gemm(stream, bufN1, Wp(10), Bf(11), N, 256, 256, 256, nullptr, bufN2, 1);
  launch_gemm(stream, bufN2, Wp(12), Bf(13), N, 256, 256, 256, xFa, xB, 0);

  // ---- edge basis + edge_mlp ----
  k_edge_geom<<<cdiv((long long)E * 64, 256), 256, 0, stream>>>(pos, lat, ei, pbc, ebat,
                                                                eAttrB, E);
  launch_gemm(stream, eAttrB, Wp(14), Bf(15), E, 64,  256, 256, nullptr, ebufA, 1);
  launch_gemm(stream, ebufA,  Wp(16), Bf(17), E, 256, 256, 256, nullptr, ebufB, 1);
  launch_gemm(stream, ebufB,  Wp(18), Bf(19), E, 256, 256, 256, nullptr, eaB,   0);

  // ---- two GATv2 layers ----
  float* xcur = xFa;
  float* xnxt = xFb;
  for (int l = 0; l < 2; ++l) {
    const int base = 20 + l * 6;   // Wl, Wr, We, att, bias, Wres
    launch_gemm(stream, xB,  Wp(base + 0), nullptr, N, 256, 256, 256, xlF,   nullptr, 0);
    launch_gemm(stream, xB,  Wp(base + 1), nullptr, N, 256, 256, 256, xrF,   nullptr, 0);
    launch_gemm(stream, xB,  Wp(base + 5), nullptr, N, 256, 256, 256, xresF, nullptr, 0);
    launch_gemm(stream, eaB, Wp(base + 2), nullptr, E, 256, 256, 256, nullptr, ebufA, 0);

    k_fill<<<cdiv((long long)N * 4, 256), 256, 0, stream>>>(mF, -INFINITY, N * 4);
    k_fill<<<cdiv((long long)N * 4, 256), 256, 0, stream>>>(denF, 0.f, N * 4);
    k_fill<<<cdiv((long long)N * 256, 256), 256, 0, stream>>>(aggF, 0.f, N * 256);

    k_gat_logit<<<cdiv((long long)E * 4, 256), 256, 0, stream>>>(
        xlF, xrF, ebufA, ei, Bf(base + 3), logitF, mF, E);
    k_gat_expsum<<<cdiv((long long)E * 4, 256), 256, 0, stream>>>(
        logitF, mF, ei, pexF, denF, E);
    k_gat_aggr<<<cdiv((long long)E * 256, 256), 256, 0, stream>>>(
        pexF, denF, xlF, ei, aggF, E);
    k_gat_combine<<<cdiv((long long)N * 256, 256), 256, 0, stream>>>(
        aggF, Bf(base + 4), xresF, xnxt, xB, N);

    float* t = xcur; xcur = xnxt; xnxt = t;
  }

  // ---- pooling ----
  launch_gemm(stream, xB,    Wp(32), Bf(33), N, 256, 256, 256, nullptr, bufN1, 1);
  launch_gemm(stream, bufN1, Wp(34), Bf(35), N, 256, 256, 256, nullptr, bufN2, 1);
  launch_gemm(stream, bufN2, Wp(36), Bf(37), N, 256, 256, 256, gF, nullptr, 0);

  k_fill<<<cdiv((long long)G * 256, 256), 256, 0, stream>>>(gmF, -INFINITY, G * 256);
  k_fill<<<cdiv((long long)G * 256, 256), 256, 0, stream>>>(gdF, 0.f, G * 256);
  k_fill<<<cdiv((long long)G * 256, 256), 256, 0, stream>>>(pooledF, 0.f, G * 256);
  k_pool_max<<<cdiv((long long)N * 256, 256), 256, 0, stream>>>(gF, nbat, gmF, N);
  k_pool_expsum<<<cdiv((long long)N * 256, 256), 256, 0, stream>>>(gF, gmF, nbat, gpF, gdF, N);
  k_pool_weighted<<<cdiv((long long)N * 256, 256), 256, 0, stream>>>(gpF, gdF, xcur, nbat,
                                                                     pooledF, N);
  k_f32_to_bf16<<<cdiv((long long)G * 256, 256), 256, 0, stream>>>(pooledF, pooledB, G * 256);

  // ---- spectra_mlp: 256 -> 2048 -> 2048 -> 4002 (f32 direct to d_out) ----
  float* epsOut = (float*)d_out;
  launch_gemm(stream, pooledB, Wp(38), Bf(39), G, 256,  2048, 2048, nullptr, sH1, 1);
  launch_gemm(stream, sH1,     Wp(40), Bf(41), G, 2048, 2048, 2048, nullptr, sH2, 1);
  launch_gemm(stream, sH2,     Wp(42), Bf(43), G, 2048, 4002, 4096, epsOut, nullptr, 0);

  // ---- drude_mlp: 256 -> 256 -> 256 -> 1, relu on final output ----
  float* drudeOut = epsOut + (size_t)G * 4002;
  launch_gemm(stream, pooledB, Wp(44), Bf(45), G, 256, 256, 256, nullptr, bufN1, 1);
  launch_gemm(stream, bufN1,   Wp(46), Bf(47), G, 256, 256, 256, nullptr, bufN2, 1);
  launch_gemm(stream, bufN2,   Wp(48), Bf(49), G, 256, 1,   256, drudeOut, nullptr, 1);
}
